// GBPN_87084756893764
// MI455X (gfx1250) — matrix operand order repaired and checked by reading, not compiled, not run on hardware
//
#include <hip/hip_runtime.h>
#include <math.h>

// ---- problem constants (match reference setup_inputs) ----
#define NN    50000
#define DIN_  512
#define DH_   128
#define CC    8
#define EE    800000
#define KITER 5

typedef __attribute__((ext_vector_type(16))) __bf16 v16bf;
typedef __attribute__((ext_vector_type(8)))  float  v8f;
typedef __attribute__((ext_vector_type(4)))  float  v4f;

// ============================================================
// W1T[c][k] = bf16(W1[k][c])   (128 x 512 bf16, L2-resident)
// ============================================================
__global__ void k_w1t(const float* __restrict__ W1, __bf16* __restrict__ W1T) {
    int i = blockIdx.x * blockDim.x + threadIdx.x;
    if (i >= DIN_ * DH_) return;
    int k = i / DH_, c = i % DH_;
    W1T[(size_t)c * DIN_ + k] = (__bf16)W1[i];
}

// ============================================================
// logH = logsigmoid(param + param^T)    (8x8)
// ============================================================
__global__ void k_logh(const float* __restrict__ p, float* __restrict__ logH) {
    int i = threadIdx.x;
    if (i >= CC * CC) return;
    int r = i / CC, c = i % CC;
    float v = p[r * CC + c] + p[c * CC + r];
    // logsigmoid(v) = min(v,0) - log1p(exp(-|v|))
    logH[i] = fminf(v, 0.0f) - log1pf(__expf(-fabsf(v)));
}

__global__ void k_zero(float* __restrict__ a, int n) {
    int i = blockIdx.x * blockDim.x + threadIdx.x;
    if (i < n) a[i] = 0.0f;
}

// ============================================================
// GEMM1: h = relu(x @ W1 + b1)    via v_wmma_f32_16x16x32_bf16
// block = 256 threads (8 waves); block computes 128 rows x 128 cols
// wave w -> rows [base+16w, base+16w+16), all 128 cols (8 16x16 tiles)
// ============================================================
__global__ void __launch_bounds__(256)
k_gemm1(const float* __restrict__ x, const __bf16* __restrict__ W1T,
        const float* __restrict__ b1, float* __restrict__ hout) {
    const int lane = threadIdx.x & 31;
    const int wv   = threadIdx.x >> 5;      // 0..7
    const int m16  = lane & 15;
    const int kh   = lane >> 4;             // K-half select (contiguous k order, same for A and B)
    const int rowbase = blockIdx.x * 128 + wv * 16;

    // A-fragment source row for this lane (lanes 0-15 and 16-31 share rows)
    int rowA = rowbase + m16;
    if (rowA >= NN) rowA = NN - 1;          // clamp; stores are guarded
    const float* xrow = x + (size_t)rowA * DIN_;

    float bias[8];
#pragma unroll
    for (int t = 0; t < 8; ++t) bias[t] = b1[16 * t + m16];

    v8f acc[8] = {};

    for (int k0 = 0; k0 < DIN_; k0 += 32) {
        // ---- A fragment (16x32 bf16): lane holds rows m16,
        //      k = {kh*16..kh*16+15}
        const v4f* pa0 = (const v4f*)(xrow + k0 + kh * 16);
        const v4f* pa1 = pa0 + 2;
        v4f a0 = pa0[0], a1 = pa0[1];
        v4f a2 = pa1[0], a3 = pa1[1];
        v16bf A;
#pragma unroll
        for (int i = 0; i < 4; ++i) {
            A[i]      = (__bf16)a0[i];
            A[4 + i]  = (__bf16)a1[i];
            A[8 + i]  = (__bf16)a2[i];
            A[12 + i] = (__bf16)a3[i];
        }
        // ---- 8 column tiles; B fragment = W1T[col][k0 + kh*16 + 0..15]
#pragma unroll
        for (int t = 0; t < 8; ++t) {
            const v16bf B = *(const v16bf*)(W1T + (size_t)(16 * t + m16) * DIN_
                                                 + k0 + kh * 16);
            acc[t] = __builtin_amdgcn_wmma_f32_16x16x32_bf16(
                false, A, false, B, (short)0, acc[t], false, false);
        }
    }

    // ---- store: VGPR r -> row (rowbase + r + 8*kh), col 16t+m16; bias+relu
#pragma unroll
    for (int t = 0; t < 8; ++t) {
        const int col = 16 * t + m16;
#pragma unroll
        for (int r = 0; r < 8; ++r) {
            int row = rowbase + r + 8 * kh;
            if (row < NN)
                hout[(size_t)row * DH_ + col] = fmaxf(acc[t][r] + bias[t], 0.0f);
        }
    }
}

// ============================================================
// GEMM2 + log_softmax: log_b0 = log_softmax(h @ W2 + b2)
// also seeds log_b (d_out) with log_b0
// ============================================================
__global__ void __launch_bounds__(256)
k_gemm2(const float* __restrict__ h, const float* __restrict__ W2,
        const float* __restrict__ b2, float* __restrict__ lb0,
        float* __restrict__ logb) {
    __shared__ float sW2[DH_ * CC];
    __shared__ float sb2[CC];
    for (int i = threadIdx.x; i < DH_ * CC; i += blockDim.x) sW2[i] = W2[i];
    if (threadIdx.x < CC) sb2[threadIdx.x] = b2[threadIdx.x];
    __syncthreads();

    int n = blockIdx.x * blockDim.x + threadIdx.x;
    if (n >= NN) return;
    const float* hr = h + (size_t)n * DH_;

    float z[CC];
#pragma unroll
    for (int c = 0; c < CC; ++c) z[c] = sb2[c];
    for (int k = 0; k < DH_; k += 4) {
        v4f hv = *(const v4f*)(hr + k);
#pragma unroll
        for (int kk = 0; kk < 4; ++kk)
#pragma unroll
            for (int c = 0; c < CC; ++c)
                z[c] = fmaf(hv[kk], sW2[(k + kk) * CC + c], z[c]);
    }
    float mx = z[0];
#pragma unroll
    for (int c = 1; c < CC; ++c) mx = fmaxf(mx, z[c]);
    float s = 0.0f;
#pragma unroll
    for (int c = 0; c < CC; ++c) s += __expf(z[c] - mx);
    float lse = mx + __logf(s);
#pragma unroll
    for (int c = 0; c < CC; ++c) {
        float v = z[c] - lse;
        lb0[(size_t)n * CC + c]  = v;
        logb[(size_t)n * CC + c] = v;
    }
}

// ============================================================
// BP message kernel: one thread per directed edge
// out[c] = LSE_j( xj[j] + w*logH[j][c] ), normalized over c
// scatter-add into agg[dst]
// ============================================================
__global__ void __launch_bounds__(256)
k_msg(const float* __restrict__ logb, const float* __restrict__ prev_msg,
      float* __restrict__ cur_msg, const int* __restrict__ src,
      const int* __restrict__ dst, const float* __restrict__ ew,
      const int* __restrict__ rv, const float* __restrict__ logH,
      float* __restrict__ agg, int use_prev) {
    __shared__ float sH[CC * CC];
    if (threadIdx.x < CC * CC) sH[threadIdx.x] = logH[threadIdx.x];
    __syncthreads();

    int e = blockIdx.x * blockDim.x + threadIdx.x;
    if (e >= EE) return;

    int s = src[e], d = dst[e];
    float w = ew[e];

    float xj[CC];
    {
        const v4f* p = (const v4f*)(logb + (size_t)s * CC);
        v4f u0 = p[0], u1 = p[1];
#pragma unroll
        for (int i = 0; i < 4; ++i) { xj[i] = u0[i]; xj[4 + i] = u1[i]; }
    }
    if (use_prev) {
        int r = rv[e];
        const v4f* p = (const v4f*)(prev_msg + (size_t)r * CC);
        v4f u0 = p[0], u1 = p[1];
#pragma unroll
        for (int i = 0; i < 4; ++i) { xj[i] -= u0[i]; xj[4 + i] -= u1[i]; }
    }

    float out[CC];
#pragma unroll
    for (int c = 0; c < CC; ++c) {
        float t[CC];
        float mx = -1e30f;
#pragma unroll
        for (int j = 0; j < CC; ++j) {
            t[j] = fmaf(w, sH[j * CC + c], xj[j]);
            mx = fmaxf(mx, t[j]);
        }
        float sum = 0.0f;
#pragma unroll
        for (int j = 0; j < CC; ++j) sum += __expf(t[j] - mx);
        out[c] = mx + __logf(sum);
    }
    // normalize over c
    float mx2 = out[0];
#pragma unroll
    for (int c = 1; c < CC; ++c) mx2 = fmaxf(mx2, out[c]);
    float s2 = 0.0f;
#pragma unroll
    for (int c = 0; c < CC; ++c) s2 += __expf(out[c] - mx2);
    float lse = mx2 + __logf(s2);

    v4f o0, o1;
#pragma unroll
    for (int i = 0; i < 4; ++i) { o0[i] = out[i] - lse; o1[i] = out[4 + i] - lse; }
    v4f* mp = (v4f*)(cur_msg + (size_t)e * CC);
    mp[0] = o0;
    mp[1] = o1;

    float* ap = agg + (size_t)d * CC;
#pragma unroll
    for (int c = 0; c < 4; ++c) atomicAdd(ap + c, o0[c]);
#pragma unroll
    for (int c = 0; c < 4; ++c) atomicAdd(ap + 4 + c, o1[c]);
}

// ============================================================
// Belief update: log_b = log_normalize(log_b0 + agg); agg := 0
// ============================================================
__global__ void __launch_bounds__(256)
k_update(const float* __restrict__ lb0, float* __restrict__ agg,
         float* __restrict__ logb) {
    int n = blockIdx.x * blockDim.x + threadIdx.x;
    if (n >= NN) return;
    float v[CC];
    v4f* ap = (v4f*)(agg + (size_t)n * CC);
    v4f a0 = ap[0], a1 = ap[1];
    const v4f* bp = (const v4f*)(lb0 + (size_t)n * CC);
    v4f b0 = bp[0], b1 = bp[1];
#pragma unroll
    for (int i = 0; i < 4; ++i) { v[i] = b0[i] + a0[i]; v[4 + i] = b1[i] + a1[i]; }
    ap[0] = (v4f)0.0f;                  // reset for next iteration / next call
    ap[1] = (v4f)0.0f;
    float mx = v[0];
#pragma unroll
    for (int c = 1; c < CC; ++c) mx = fmaxf(mx, v[c]);
    float s = 0.0f;
#pragma unroll
    for (int c = 0; c < CC; ++c) s += __expf(v[c] - mx);
    float lse = mx + __logf(s);
    v4f* op = (v4f*)(logb + (size_t)n * CC);
    v4f o0, o1;
#pragma unroll
    for (int i = 0; i < 4; ++i) { o0[i] = v[i] - lse; o1[i] = v[4 + i] - lse; }
    op[0] = o0;
    op[1] = o1;
}

// ============================================================
// launch
// ============================================================
extern "C" void kernel_launch(void* const* d_in, const int* in_sizes, int n_in,
                              void* d_out, int out_size, void* d_ws, size_t ws_size,
                              hipStream_t stream) {
    const float* x   = (const float*)d_in[0];
    const int*   ei  = (const int*)  d_in[1];   // [2, E]
    const float* ew  = (const float*)d_in[2];
    const int*   rv  = (const int*)  d_in[3];
    const float* W1  = (const float*)d_in[4];
    const float* b1  = (const float*)d_in[5];
    const float* W2  = (const float*)d_in[6];
    const float* b2  = (const float*)d_in[7];
    const float* par = (const float*)d_in[8];
    const int* srcp = ei;
    const int* dstp = ei + EE;

    // workspace layout (256B-aligned offsets), total ~80.2 MB
    char* ws = (char*)d_ws;
    __bf16* W1T  = (__bf16*)(ws + 0);            //   131072 B
    float*  h    = (float*)(ws + 131072);        // 25600000 B
    float*  lb0  = (float*)(ws + 25731072);      //  1600000 B
    float*  agg  = (float*)(ws + 27331072);      //  1600000 B
    float*  logH = (float*)(ws + 28931072);      //      256 B
    float*  msg0 = (float*)(ws + 28931328);      // 25600000 B
    float*  msg1 = (float*)(ws + 54531328);      // 25600000 B
    float*  logb = (float*)d_out;                // live beliefs [N, C]

    k_w1t <<<(DIN_ * DH_ + 255) / 256, 256, 0, stream>>>(W1, W1T);
    k_logh<<<1, 64, 0, stream>>>(par, logH);
    k_zero<<<(NN * CC + 255) / 256, 256, 0, stream>>>(agg, NN * CC);

    k_gemm1<<<(NN + 127) / 128, 256, 0, stream>>>(x, W1T, b1, h);
    k_gemm2<<<(NN + 255) / 256, 256, 0, stream>>>(h, W2, b2, lb0, logb);

    for (int it = 0; it < KITER; ++it) {
        float* cur  = (it & 1) ? msg1 : msg0;
        float* prev = (it & 1) ? msg0 : msg1;   // unread when it == 0
        k_msg<<<(EE + 255) / 256, 256, 0, stream>>>(
            logb, prev, cur, srcp, dstp, ew, rv, logH, agg, it > 0 ? 1 : 0);
        k_update<<<(NN + 255) / 256, 256, 0, stream>>>(lb0, agg, logb);
    }
}
